// TernaryConv2d_42597485641953
// MI455X (gfx1250) — matrix operand order repaired and checked by reading, not compiled, not run on hardware
//
#include <hip/hip_runtime.h>
#include <hip/hip_bf16.h>

typedef __attribute__((ext_vector_type(16))) _Float16 v16h;
typedef __attribute__((ext_vector_type(8)))  _Float16 v8h;
typedef __attribute__((ext_vector_type(8)))  float    v8f;

// Problem constants (from reference)
#define NB      32
#define CIN     128
#define HIN     56
#define WIN     56
#define OC      256
#define OH      54
#define OW      54
#define KTOT    1152            // 128*3*3
#define PIX_IMG 2916            // 54*54
#define NPIX    93312           // 32*2916
#define NW      294912          // 256*128*3*3
#define CHW     (CIN * HIN * WIN)
#define HW      (HIN * WIN)

// Tiling
#define OC_TILE 128             // workgroup M tile (channels)
#define PX_TILE 128             // workgroup N tile (pixels)
#define K_STEP  32
#define NKSTEP  (KTOT / K_STEP) // 36

union FragA  { v16h v; v8h h[2]; };
union Pack8  { v8h v; _Float16 h[8]; };

// ---------------- kernel 1: max |w| reduction ----------------
__global__ void twn_max_kernel(const float* __restrict__ w, unsigned* __restrict__ out) {
    __shared__ float red[256];
    float m = 0.0f;
    for (int i = blockIdx.x * blockDim.x + threadIdx.x; i < NW; i += gridDim.x * blockDim.x)
        m = fmaxf(m, fabsf(w[i]));
    red[threadIdx.x] = m;
    __syncthreads();
    for (int s = 128; s > 0; s >>= 1) {
        if (threadIdx.x < (unsigned)s) red[threadIdx.x] = fmaxf(red[threadIdx.x], red[threadIdx.x + s]);
        __syncthreads();
    }
    if (threadIdx.x == 0) atomicMax(out, __float_as_uint(red[0]));  // bits monotonic for w >= 0
}

// ---------------- kernel 2: ternary quantize + permute to f16 [oc][rs*128+c] ----------------
__global__ void twn_quant_kernel(const float* __restrict__ w, const unsigned* __restrict__ maxbits,
                                 _Float16* __restrict__ wq) {
    int i = blockIdx.x * blockDim.x + threadIdx.x;
    if (i >= NW) return;
    float t = 0.05f * __uint_as_float(*maxbits);
    float v = w[i];
    float q = (v > t) ? 1.0f : ((v < -t) ? -1.0f : 0.0f);
    // input layout OIHW flat: i = ((oc*128 + c)*3 + r)*3 + s
    int oc  = i / KTOT;
    int rem = i - oc * KTOT;
    int c   = rem / 9;
    int rs  = rem - 9 * c;      // r*3+s
    wq[(size_t)oc * KTOT + rs * CIN + c] = (_Float16)q;
}

// ---------------- kernel 3: implicit-GEMM conv via WMMA f16 ----------------
// A (MxK) = ternary weights, B (KxN) = im2col activations; K' = (r*3+s)*128 + c.
// Workgroup tile 128ch x 128px, 8 waves as 4(M) x 2(N), wave tile 32x64 (2x4 WMMA).
__global__ void __launch_bounds__(256, 2)
twn_conv_wmma_kernel(const float* __restrict__ x, const _Float16* __restrict__ wq,
                     const float* __restrict__ bias, float* __restrict__ out) {
    __shared__ __align__(32) _Float16 sW[2][OC_TILE * K_STEP];   // 2 x 8 KB
    __shared__ __align__(32) _Float16 sX[2][PX_TILE * K_STEP];   // 2 x 8 KB

    const int tid   = threadIdx.x;
    const int lane  = tid & 31;
    const int wave  = tid >> 5;
    const int waveM = wave >> 1;          // 0..3 -> channel sub-tile of 32
    const int waveN = wave & 1;           // 0..1 -> pixel sub-tile of 64
    const int lmod  = lane & 15;
    const int lhalf = lane >> 4;          // 0: lanes 0-15, 1: lanes 16-31

    const int pxBase = blockIdx.x * PX_TILE;   // 729 tiles, exact
    const int ocBase = blockIdx.y * OC_TILE;   // 0 or 128

    // ---- X staging role: lane axis = pixels (coalesced), thread covers 16 contiguous K ----
    const int pxl = tid & 127;            // pixel handled by this thread
    const int kr  = tid >> 7;             // 0..1 -> K sub-range [kr*16, kr*16+16)
    int pbase;
    {
        int p  = pxBase + pxl;
        int nb = p / PIX_IMG;
        int q  = p - nb * PIX_IMG;
        int oh = q / OW;
        int ow = q - oh * OW;
        pbase = nb * CHW + oh * WIN + ow;         // + c*3136 + r*56 + s at use
    }

    // ---- W staging role: thread copies 16 halves of row (tid>>1) at half-offset (tid&1)*16
    const int wrow = tid >> 1;
    const int woff = (tid & 1) * 16;
    const _Float16* wsrc_base = wq + (size_t)(ocBase + wrow) * KTOT + woff;

    v8f acc[2][4];
    #pragma unroll
    for (int i = 0; i < 2; ++i)
        #pragma unroll
        for (int j = 0; j < 4; ++j)
            acc[i][j] = (v8f){0.f, 0.f, 0.f, 0.f, 0.f, 0.f, 0.f, 0.f};

    auto stage = [&](int kt, int buf) {
        // weights: 2 x 16-byte vector loads -> LDS
        const v8h* wsrc = (const v8h*)(wsrc_base + kt * K_STEP);
        v8h* wd = (v8h*)&sW[buf][wrow * K_STEP + woff];
        wd[0] = wsrc[0];
        wd[1] = wsrc[1];
        // activations: fixed (r,s) per K-step, 32 consecutive channels
        int rs    = kt >> 2;              // 0..8
        int r     = (rs * 11) >> 5;       // exact /3 for rs<9
        int s     = rs - 3 * r;
        int cbase = (kt & 3) * 32 + kr * 16;
        const float* xp = x + pbase + r * WIN + s + cbase * HW;
        Pack8 p0, p1;
        #pragma unroll
        for (int i = 0; i < 8; ++i) {
            p0.h[i] = (_Float16)xp[i * HW];          // lanes coalesced over pixels
            p1.h[i] = (_Float16)xp[(i + 8) * HW];
        }
        v8h* xd = (v8h*)&sX[buf][pxl * K_STEP + kr * 16];
        xd[0] = p0.v;                                 // 2 x ds_store_b128
        xd[1] = p1.v;
    };

    auto compute = [&](int buf) {
        FragA a[2];
        #pragma unroll
        for (int i = 0; i < 2; ++i) {
            const v8h* row = (const v8h*)&sW[buf][(waveM * 32 + i * 16 + lmod) * K_STEP];
            a[i].h[0] = row[lhalf * 2];      // K = lhalf*16 .. +8
            a[i].h[1] = row[lhalf * 2 + 1];  // K = lhalf*16+8 .. +8
        }
        FragA bf[4];
        #pragma unroll
        for (int j = 0; j < 4; ++j) {
            const v8h* row = (const v8h*)&sX[buf][(waveN * 64 + j * 16 + lmod) * K_STEP];
            bf[j].h[0] = row[lhalf * 2];     // K = lhalf*16 .. +16 contiguous
            bf[j].h[1] = row[lhalf * 2 + 1];
        }
        #pragma unroll
        for (int i = 0; i < 2; ++i)
            #pragma unroll
            for (int j = 0; j < 4; ++j)
                acc[i][j] = __builtin_amdgcn_wmma_f32_16x16x32_f16(
                    false, a[i].v, false, bf[j].v, (short)0, acc[i][j], false, false);
    };

    stage(0, 0);
    for (int kt = 0; kt < NKSTEP; kt += 2) {
        __syncthreads();
        stage(kt + 1, 1);                 // overlap next-tile loads with WMMA
        compute(0);
        __syncthreads();
        if (kt + 2 < NKSTEP) stage(kt + 2, 0);
        compute(1);
    }

    // ---- epilogue: bias + coalesced scattered store ----
    const int ocb = ocBase + waveM * 32;
    float bv[2][8];
    #pragma unroll
    for (int i = 0; i < 2; ++i)
        #pragma unroll
        for (int v = 0; v < 8; ++v)
            bv[i][v] = bias[ocb + i * 16 + v + lhalf * 8];

    #pragma unroll
    for (int j = 0; j < 4; ++j) {
        int px = pxBase + waveN * 64 + j * 16 + lmod;    // pixel for this lane (fixed over VGPRs)
        int nb = px / PIX_IMG;
        int q  = px - nb * PIX_IMG;
        float* op = out + (size_t)nb * OC * PIX_IMG + q;
        #pragma unroll
        for (int i = 0; i < 2; ++i) {
            #pragma unroll
            for (int v = 0; v < 8; ++v) {
                int oc = ocb + i * 16 + v + lhalf * 8;
                op[(size_t)oc * PIX_IMG] = acc[i][j][v] + bv[i][v];
            }
        }
    }
}

extern "C" void kernel_launch(void* const* d_in, const int* in_sizes, int n_in,
                              void* d_out, int out_size, void* d_ws, size_t ws_size,
                              hipStream_t stream) {
    const float* x    = (const float*)d_in[0];
    const float* w    = (const float*)d_in[1];
    const float* bias = (const float*)d_in[2];
    float*       out  = (float*)d_out;

    unsigned*  maxbits = (unsigned*)d_ws;
    _Float16*  wq      = (_Float16*)((char*)d_ws + 256);   // 256B-aligned, ~590 KB

    hipMemsetAsync(d_ws, 0, 4, stream);

    twn_max_kernel<<<256, 256, 0, stream>>>(w, maxbits);
    twn_quant_kernel<<<(NW + 255) / 256, 256, 0, stream>>>(w, maxbits, wq);

    dim3 grid(NPIX / PX_TILE, OC / OC_TILE);   // (729, 2)
    twn_conv_wmma_kernel<<<grid, 256, 0, stream>>>(x, wq, bias, out);
}
